// CRFClassifier_8220567405029
// MI455X (gfx1250) — matrix operand order repaired and checked by reading, not compiled, not run on hardware
//
#include <hip/hip_runtime.h>
#include <hip/hip_bf16.h>
#include <stdint.h>

#define B_    64
#define T_    8192
#define D_    128
#define K_    9
#define H1_   128
#define H2_   64
#define CHUNK 128
#define NC    (T_ / CHUNK)   // 64 chunks per batch

typedef __attribute__((ext_vector_type(16))) _Float16 v16h;
typedef __attribute__((ext_vector_type(8)))  float    v8f;

union Frag { v16h v; uint32_t u[8]; };

__device__ __forceinline__ uint32_t pkh(float a, float b) {
  union { _Float16 h[2]; uint32_t u; } t;
  t.h[0] = (_Float16)a; t.h[1] = (_Float16)b;
  return t.u;
}

// ---------------- workspace layout (bytes) ----------------
static constexpr size_t OFF_EM  = 0;                                    // B*T*K f32 emissions
static constexpr size_t OFF_W1  = OFF_EM  + (size_t)B_*T_*K_*4;         // 128x128 f16 packed
static constexpr size_t OFF_W2  = OFF_W1  + (size_t)D_*H1_*2;           // 128x64 f16 packed
static constexpr size_t OFF_W3  = OFF_W2  + (size_t)H1_*H2_*2;          // 64x16 f16 packed (9->16 pad)
static constexpr size_t OFF_CM  = OFF_W3  + (size_t)H2_*16*2;           // B*NC*81 f32 chunk matrices
static constexpr size_t OFF_CS  = OFF_CM  + (size_t)B_*NC*81*4;         // B*(NC+1)*16 f32 chunk-entry scores
static constexpr size_t OFF_BP  = OFF_CS  + (size_t)B_*(NC+1)*16*4;     // B*T*12 bytes backpointers
static constexpr size_t OFF_CG  = OFF_BP  + (size_t)B_*T_*12;           // B*NC*16 bytes chunk maps
static constexpr size_t OFF_CET = OFF_CG  + (size_t)B_*NC*16;           // B*NC int32 chunk-end tags

// ---------------- weight packing into WMMA B-fragment layout ----------------
// B frag (32x16 f16): VGPR p, k = k0 + 8*grp + 16*(p>>2) + 2*(p&3) (+1);
// grp = lane>>4. Packed so each lane reads 8 consecutive dwords.
__device__ __forceinline__ void wb_fill(uint32_t* dst, const float* W, int N, int Nsrc, int idx) {
  int p    = idx & 7;
  int lane = (idx >> 3) & 31;
  int blk  = idx >> 8;
  int nblkCount = N >> 4;
  int kblk = blk / nblkCount;
  int nblk = blk - kblk * nblkCount;
  int grp = lane >> 4, ln = lane & 15;
  int n = nblk * 16 + ln;
  int k = kblk * 32 + grp * 8 + (p >> 2) * 16 + 2 * (p & 3);
  float a = (n < Nsrc) ? W[(size_t)k * Nsrc + n]       : 0.f;
  float b = (n < Nsrc) ? W[(size_t)(k + 1) * Nsrc + n] : 0.f;
  dst[idx] = pkh(a, b);
}

__global__ __launch_bounds__(256) void k_prep_wb(const float* __restrict__ W1,
                                                 const float* __restrict__ W2,
                                                 const float* __restrict__ W3,
                                                 uint32_t* wb1, uint32_t* wb2, uint32_t* wb3) {
  int tid = blockIdx.x * blockDim.x + threadIdx.x;
  const int S1 = (D_ / 32) * (H1_ / 16) * 256;   // 8192 dwords
  const int S2 = (H1_ / 32) * (H2_ / 16) * 256;  // 4096 dwords
  const int S3 = (H2_ / 32) * 1 * 256;           // 512 dwords
  if (tid < S1)                wb_fill(wb1, W1, H1_, H1_, tid);
  else if (tid < S1 + S2)      wb_fill(wb2, W2, H2_, H2_, tid - S1);
  else if (tid < S1 + S2 + S3) wb_fill(wb3, W3, 16, K_, tid - S1 - S2);
}

// ---------------- fragment loaders ----------------
__device__ __forceinline__ Frag load_a_global(const float* __restrict__ row, int k0, int grp) {
  Frag a;
  const float4* q0 = (const float4*)(row + k0 + grp * 8);
  const float4* q1 = (const float4*)(row + k0 + 16 + grp * 8);
  float4 f0 = q0[0], f1 = q0[1], f2 = q1[0], f3 = q1[1];
  a.u[0] = pkh(f0.x, f0.y); a.u[1] = pkh(f0.z, f0.w);
  a.u[2] = pkh(f1.x, f1.y); a.u[3] = pkh(f1.z, f1.w);
  a.u[4] = pkh(f2.x, f2.y); a.u[5] = pkh(f2.z, f2.w);
  a.u[6] = pkh(f3.x, f3.y); a.u[7] = pkh(f3.z, f3.w);
  return a;
}

__device__ __forceinline__ Frag load_a_lds(const _Float16* h, int stride, int row, int k0, int grp) {
  Frag a;
  const uint4* q0 = (const uint4*)(h + row * stride + k0 + grp * 8);
  const uint4* q1 = (const uint4*)(h + row * stride + k0 + 16 + grp * 8);
  uint4 a0 = *q0, a1 = *q1;
  a.u[0] = a0.x; a.u[1] = a0.y; a.u[2] = a0.z; a.u[3] = a0.w;
  a.u[4] = a1.x; a.u[5] = a1.y; a.u[6] = a1.z; a.u[7] = a1.w;
  return a;
}

__device__ __forceinline__ Frag load_b(const uint32_t* __restrict__ wb, int nblkCount,
                                       int kblk, int nblk, int lane) {
  Frag b;
  const uint4* q = (const uint4*)(wb + ((size_t)((kblk * nblkCount + nblk) * 32 + lane)) * 8);
  uint4 q0 = q[0], q1 = q[1];
  b.u[0] = q0.x; b.u[1] = q0.y; b.u[2] = q0.z; b.u[3] = q0.w;
  b.u[4] = q1.x; b.u[5] = q1.y; b.u[6] = q1.z; b.u[7] = q1.w;
  return b;
}

#define WMMA_F16(A, Bf, C) \
  __builtin_amdgcn_wmma_f32_16x16x32_f16(false, (A), false, (Bf), (short)0, (C), false, false)

// ---------------- MLP: 64 tokens / block, 8 waves ----------------
__global__ __launch_bounds__(256) void k_mlp(const float* __restrict__ x,
                                             const float* __restrict__ b1,
                                             const float* __restrict__ b2,
                                             const float* __restrict__ b3,
                                             const uint32_t* __restrict__ wb1,
                                             const uint32_t* __restrict__ wb2,
                                             const uint32_t* __restrict__ wb3,
                                             float* __restrict__ em) {
  __shared__ __align__(16) _Float16 h1[64 * 136];  // stride 136 (16B-aligned rows)
  __shared__ __align__(16) _Float16 h2[64 * 72];   // stride 72
  const int w = threadIdx.x >> 5, lane = threadIdx.x & 31;
  const int grp = lane >> 4, ln = lane & 15;
  const long tokBase = (long)blockIdx.x * 64;

  if (blockIdx.x + 1 < gridDim.x)
    __builtin_prefetch(x + (tokBase + 64) * D_, 0, 1);  // global_prefetch_b8 next tile

  // ---- Layer 1: M=64, N=128, K=128; wave w owns column block n0=w*16
  {
    const int n0 = w * 16;
    Frag bfr[4];
#pragma unroll
    for (int kb = 0; kb < 4; ++kb) bfr[kb] = load_b(wb1, H1_ / 16, kb, w, lane);
    float bias = b1[n0 + ln];
#pragma unroll
    for (int mt = 0; mt < 4; ++mt) {
      v8f acc = {};
#pragma unroll
      for (int kb = 0; kb < 4; ++kb) {
        Frag a = load_a_global(x + (tokBase + mt * 16 + ln) * D_, kb * 32, grp);
        acc = WMMA_F16(a.v, bfr[kb].v, acc);
      }
#pragma unroll
      for (int r = 0; r < 8; ++r) {
        float v = acc[r] + bias; v = v > 0.f ? v : 0.f;
        h1[(mt * 16 + r + grp * 8) * 136 + n0 + ln] = (_Float16)v;
      }
    }
  }
  __syncthreads();

  // ---- Layer 2: M=64, N=64, K=128; wave w: n0=(w&3)*16, mtiles (w>>2)*2..+1
  {
    const int n0 = (w & 3) * 16;
    Frag bfr[4];
#pragma unroll
    for (int kb = 0; kb < 4; ++kb) bfr[kb] = load_b(wb2, H2_ / 16, kb, w & 3, lane);
    float bias = b2[n0 + ln];
#pragma unroll
    for (int m2 = 0; m2 < 2; ++m2) {
      const int mt = (w >> 2) * 2 + m2;
      v8f acc = {};
#pragma unroll
      for (int kb = 0; kb < 4; ++kb) {
        Frag a = load_a_lds(h1, 136, mt * 16 + ln, kb * 32, grp);
        acc = WMMA_F16(a.v, bfr[kb].v, acc);
      }
#pragma unroll
      for (int r = 0; r < 8; ++r) {
        float v = acc[r] + bias; v = v > 0.f ? v : 0.f;
        h2[(mt * 16 + r + grp * 8) * 72 + n0 + ln] = (_Float16)v;
      }
    }
  }
  __syncthreads();

  // ---- Layer 3: M=64, N=16 (K=9 padded), K=64; waves 0..3, mtile=w
  if (w < 4) {
    Frag bf0 = load_b(wb3, 1, 0, 0, lane);
    Frag bf1 = load_b(wb3, 1, 1, 0, lane);
    v8f acc = {};
    Frag a0 = load_a_lds(h2, 72, w * 16 + ln, 0, grp);
    acc = WMMA_F16(a0.v, bf0.v, acc);
    Frag a1 = load_a_lds(h2, 72, w * 16 + ln, 32, grp);
    acc = WMMA_F16(a1.v, bf1.v, acc);
    if (ln < K_) {
      float bias = b3[ln];
#pragma unroll
      for (int r = 0; r < 8; ++r) {
        long tok = tokBase + w * 16 + r + grp * 8;
        em[tok * K_ + ln] = acc[r] + bias;
      }
    }
  }
}

// ---------------- Viterbi stage 1: per-chunk max-plus matrix product ----------------
// lane i holds row i of R (9 floats); trans fully unrolled -> uniform scalar loads.
__global__ __launch_bounds__(32) void k_chunk_fwd(const float* __restrict__ em,
                                                  const float* __restrict__ trans,
                                                  float* __restrict__ cm) {
  const int b = blockIdx.x / NC, c = blockIdx.x % NC;
  const int lane = threadIdx.x;
  float tr[81];
#pragma unroll
  for (int i = 0; i < 81; ++i) tr[i] = trans[i];
  float r[9];
#pragma unroll
  for (int k = 0; k < 9; ++k) r[k] = (k == lane) ? 0.f : -1.0e30f;
  const float* emb = em + (size_t)b * T_ * K_;
  const int t0 = (c == 0) ? 1 : c * CHUNK;
  const int t1 = (c + 1) * CHUNK;
  for (int t = t0; t < t1; ++t) {
    float e[9];
#pragma unroll
    for (int j = 0; j < 9; ++j) e[j] = emb[(size_t)t * K_ + j];
    float nw[9];
#pragma unroll
    for (int j = 0; j < 9; ++j) {
      float best = r[0] + tr[j];
#pragma unroll
      for (int i = 1; i < 9; ++i) { float v = r[i] + tr[i * 9 + j]; best = v > best ? v : best; }
      nw[j] = best + e[j];
    }
#pragma unroll
    for (int j = 0; j < 9; ++j) r[j] = nw[j];
  }
  if (lane < 9) {
    float* out = cm + ((size_t)(b * NC + c)) * 81 + lane * 9;
#pragma unroll
    for (int j = 0; j < 9; ++j) out[j] = r[j];
  }
}

// ---------------- Viterbi stage 2: scan chunk matrices -> chunk entry scores ----------------
__global__ __launch_bounds__(32) void k_chunk_scan(const float* __restrict__ em,
                                                   const float* __restrict__ start_t,
                                                   const float* __restrict__ cm,
                                                   float* __restrict__ cs) {
  const int b = blockIdx.x;
  float s[9];
#pragma unroll
  for (int j = 0; j < 9; ++j) s[j] = start_t[j] + em[(size_t)b * T_ * K_ + j];
  float* csb = cs + (size_t)b * (NC + 1) * 16;
  if (threadIdx.x == 0) {
#pragma unroll
    for (int j = 0; j < 9; ++j) csb[j] = s[j];
  }
  for (int c = 0; c < NC; ++c) {
    const float* A = cm + ((size_t)(b * NC + c)) * 81;
    float nw[9];
#pragma unroll
    for (int j = 0; j < 9; ++j) {
      float best = s[0] + A[j];
#pragma unroll
      for (int i = 1; i < 9; ++i) { float v = s[i] + A[i * 9 + j]; best = v > best ? v : best; }
      nw[j] = best;
    }
#pragma unroll
    for (int j = 0; j < 9; ++j) s[j] = nw[j];
    if (threadIdx.x == 0) {
#pragma unroll
      for (int j = 0; j < 9; ++j) csb[(c + 1) * 16 + j] = s[j];
    }
  }
}

// ---------------- Viterbi stage 3: backpointers + chunk backtrace map ----------------
__global__ __launch_bounds__(32) void k_chunk_bp(const float* __restrict__ em,
                                                 const float* __restrict__ trans,
                                                 const float* __restrict__ end_t,
                                                 const float* __restrict__ cs,
                                                 uint32_t* __restrict__ bp,
                                                 uint8_t* __restrict__ cg,
                                                 int* __restrict__ cet,
                                                 int* __restrict__ out) {
  __shared__ uint32_t bpl[CHUNK * 3];
  const int b = blockIdx.x / NC, c = blockIdx.x % NC;
  const int lane = threadIdx.x;
  float tr[81];
#pragma unroll
  for (int i = 0; i < 81; ++i) tr[i] = trans[i];
  float s[9];
  const float* csb = cs + (size_t)b * (NC + 1) * 16 + (size_t)c * 16;
#pragma unroll
  for (int j = 0; j < 9; ++j) s[j] = csb[j];
  const float* emb = em + (size_t)b * T_ * K_;
  uint32_t* bpg = bp + ((size_t)b * T_ + (size_t)c * CHUNK) * 3;
  const int t0 = (c == 0) ? 1 : c * CHUNK;
  const int t1 = (c + 1) * CHUNK;
  for (int t = t0; t < t1; ++t) {
    float e[9];
#pragma unroll
    for (int j = 0; j < 9; ++j) e[j] = emb[(size_t)t * K_ + j];
    float nw[9]; int bi[9];
#pragma unroll
    for (int j = 0; j < 9; ++j) {
      float best = s[0] + tr[j]; int bidx = 0;
#pragma unroll
      for (int i = 1; i < 9; ++i) {
        float v = s[i] + tr[i * 9 + j];
        if (v > best) { best = v; bidx = i; }
      }
      nw[j] = best + e[j]; bi[j] = bidx;
    }
#pragma unroll
    for (int j = 0; j < 9; ++j) s[j] = nw[j];
    if (lane == 0) {
      uint32_t w0 = (uint32_t)bi[0] | ((uint32_t)bi[1] << 8) | ((uint32_t)bi[2] << 16) | ((uint32_t)bi[3] << 24);
      uint32_t w1 = (uint32_t)bi[4] | ((uint32_t)bi[5] << 8) | ((uint32_t)bi[6] << 16) | ((uint32_t)bi[7] << 24);
      uint32_t w2 = (uint32_t)bi[8];
      int slot = t - c * CHUNK;
      bpg[(size_t)slot * 3 + 0] = w0; bpg[(size_t)slot * 3 + 1] = w1; bpg[(size_t)slot * 3 + 2] = w2;
      bpl[slot * 3 + 0] = w0; bpl[slot * 3 + 1] = w1; bpl[slot * 3 + 2] = w2;
    }
  }
  if (c == NC - 1 && lane == 0) {
    float best = s[0] + end_t[0]; int bidx = 0;
#pragma unroll
    for (int j = 1; j < 9; ++j) { float v = s[j] + end_t[j]; if (v > best) { best = v; bidx = j; } }
    cet[b * NC + NC - 1] = bidx;
    out[(size_t)b * T_ + T_ - 1] = bidx;
  }
  __syncthreads();
  // compose chunk backtrace map G_c[j] = tag at (t0-1) given tag at (t1-1) == j
  int v = lane < 9 ? lane : 0;
  const uint8_t* bpb = (const uint8_t*)bpl;
  for (int t = t1 - 1; t >= t0; --t) v = bpb[(t - c * CHUNK) * 12 + v];
  if (lane < 9) cg[(size_t)(b * NC + c) * 16 + lane] = (uint8_t)v;
}

// ---------------- Viterbi stage 4: chunk-level backtrace ----------------
__global__ __launch_bounds__(32) void k_chain(const uint8_t* __restrict__ cg,
                                              int* __restrict__ cet) {
  if (threadIdx.x != 0) return;
  const int b = blockIdx.x;
  int e = cet[b * NC + NC - 1];
  for (int c = NC - 1; c >= 1; --c) {
    e = cg[(size_t)(b * NC + c) * 16 + e];
    cet[b * NC + c - 1] = e;
  }
}

// ---------------- Viterbi stage 5: emit tags (within-chunk walk in LDS) ----------------
__global__ __launch_bounds__(32) void k_emit(const uint32_t* __restrict__ bp,
                                             const int* __restrict__ cet,
                                             int* __restrict__ out) {
  __shared__ uint32_t bpl[CHUNK * 3];
  const int b = blockIdx.x / NC, c = blockIdx.x % NC;
  const int lane = threadIdx.x;
  const uint32_t* bpg = bp + ((size_t)b * T_ + (size_t)c * CHUNK) * 3;
#pragma unroll
  for (int q = 0; q < 12; ++q) bpl[lane + 32 * q] = bpg[lane + 32 * q];
  __syncthreads();
  if (lane != 0) return;
  const int t0 = (c == 0) ? 1 : c * CHUNK;
  const int t1 = (c + 1) * CHUNK;
  int v = cet[b * NC + c];
  int* ob = out + (size_t)b * T_;
  ob[t1 - 1] = v;
  const uint8_t* bpb = (const uint8_t*)bpl;
  for (int t = t1 - 1; t >= t0; --t) {
    v = bpb[(t - c * CHUNK) * 12 + v];
    ob[t - 1] = v;
  }
}

// ---------------- launcher ----------------
extern "C" void kernel_launch(void* const* d_in, const int* in_sizes, int n_in,
                              void* d_out, int out_size, void* d_ws, size_t ws_size,
                              hipStream_t stream) {
  const float* x  = (const float*)d_in[0];
  const float* W1 = (const float*)d_in[1];
  const float* b1 = (const float*)d_in[2];
  const float* W2 = (const float*)d_in[3];
  const float* b2 = (const float*)d_in[4];
  const float* W3 = (const float*)d_in[5];
  const float* b3 = (const float*)d_in[6];
  const float* st = (const float*)d_in[7];
  const float* en = (const float*)d_in[8];
  const float* tr = (const float*)d_in[9];

  char* ws = (char*)d_ws;
  float*    em  = (float*)(ws + OFF_EM);
  uint32_t* wb1 = (uint32_t*)(ws + OFF_W1);
  uint32_t* wb2 = (uint32_t*)(ws + OFF_W2);
  uint32_t* wb3 = (uint32_t*)(ws + OFF_W3);
  float*    cm  = (float*)(ws + OFF_CM);
  float*    cs  = (float*)(ws + OFF_CS);
  uint32_t* bp  = (uint32_t*)(ws + OFF_BP);
  uint8_t*  cg  = (uint8_t*)(ws + OFF_CG);
  int*      cet = (int*)(ws + OFF_CET);
  int*      out = (int*)d_out;

  k_prep_wb<<<50, 256, 0, stream>>>(W1, W2, W3, wb1, wb2, wb3);
  k_mlp<<<(B_ * T_) / 64, 256, 0, stream>>>(x, b1, b2, b3, wb1, wb2, wb3, em);
  k_chunk_fwd<<<B_ * NC, 32, 0, stream>>>(em, tr, cm);
  k_chunk_scan<<<B_, 32, 0, stream>>>(em, st, cm, cs);
  k_chunk_bp<<<B_ * NC, 32, 0, stream>>>(em, tr, en, cs, bp, cg, cet, out);
  k_chain<<<B_, 32, 0, stream>>>(cg, cet);
  k_emit<<<B_ * NC, 32, 0, stream>>>(bp, cet, out);
}
